// BiLSTM_CRF_9904194584843
// MI455X (gfx1250) — matrix operand order repaired and checked by reading, compiled, not run on hardware
//
#include <hip/hip_runtime.h>
#include <hip/hip_bf16.h>

// ---------------- types ----------------
typedef __attribute__((ext_vector_type(16))) _Float16 v16h;
typedef __attribute__((ext_vector_type(8)))  _Float16 v8h;
typedef __attribute__((ext_vector_type(8)))  float    v8f;

// ---------------- model constants ----------------
constexpr int Bc  = 128;    // batch
constexpr int Tc  = 512;    // time steps
constexpr int Ec  = 256;    // embed dim
constexpr int HDc = 256;    // hidden per direction
constexpr int Hc  = 512;    // 2*HDc
constexpr int Kc  = 12;     // tags
constexpr int BEGINt = 10;
constexpr int ENDt   = 11;

// branch-free fast activations (v_exp_f32 + v_rcp_f32, no libm divergence)
__device__ __forceinline__ float sigf(float x) {
    return __builtin_amdgcn_rcpf(1.f + __expf(-x));
}
__device__ __forceinline__ float tanh_fast(float x) {
    return 1.f - 2.f * __builtin_amdgcn_rcpf(1.f + __expf(2.f * x));
}

// ---------------- 1) embedding gather + fp16 cast ----------------
__global__ void __launch_bounds__(256)
k_embed(const int* __restrict__ sent, const float* __restrict__ embed,
        _Float16* __restrict__ x16)
{
    const size_t token = blockIdx.x;            // B*T tokens
    const int    e     = threadIdx.x;           // 256 features
    const int    id    = sent[token];
    x16[token * Ec + e] = (_Float16)embed[(size_t)id * Ec + e];
}

// ---------------- 2) weight prep: Wcat = fp16([W_ih | W_hh]), bias = b_ih+b_hh ----------------
__global__ void __launch_bounds__(256)
k_prepw(const float* __restrict__ w_ih, const float* __restrict__ w_hh,
        const float* __restrict__ b_ih, const float* __restrict__ b_hh,
        int din, _Float16* __restrict__ Wc, float* __restrict__ bs)
{
    const int n  = blockIdx.x;          // 0..1023 gate rows
    const int Kd = din + HDc;
    for (int k = threadIdx.x; k < Kd; k += blockDim.x) {
        float v = (k < din) ? w_ih[(size_t)n * din + k]
                            : w_hh[(size_t)n * HDc + (k - din)];
        Wc[(size_t)n * Kd + k] = (_Float16)v;
    }
    if (threadIdx.x == 0) bs[n] = b_ih[n] + b_hh[n];
}

// ---------------- 3) persistent BiLSTM recurrence (one block per direction) ----------------
// x   : (B, T, din) fp16      Wcat: [dir][1024][din+256] fp16
// bias: [dir][1024] f32       out : (B, T, 512) fp16, direction writes cols [dir*256, ...)
__global__ void __launch_bounds__(1024)
k_lstm(const _Float16* __restrict__ x, int din,
       const _Float16* __restrict__ Wcat, const float* __restrict__ bias,
       const int* __restrict__ seq_len, _Float16* __restrict__ out)
{
    const int dir = blockIdx.x;
    const int Kd  = din + HDc;
    const _Float16* W  = Wcat + (size_t)dir * 1024 * Kd;
    const float*    bs = bias + dir * 1024;

    __shared__ _Float16 hbuf[2][Bc][HDc];   // double-buffered hidden state (128 KB)
    __shared__ float    cbuf[Bc][HDc];      // cell state (128 KB) - lane-exclusive RMW
    __shared__ int      Ls[Bc];

    const int tid   = threadIdx.x;
    const int wave  = tid >> 5;          // 0..31
    const int lane  = tid & 31;
    const int mtile = wave >> 2;         // 0..7  -> batch rows mtile*16..+15
    const int hgrp  = wave & 3;          // 0..3  -> h col-tiles hgrp*4..+3
    const int lrow  = (lane >> 4) << 3;  // C/D: M = lrow + r
    const int lcol  = lane & 15;         // C/D: N = lcol ; B: col ; A: row

    for (int i = tid; i < Bc * HDc; i += 1024) {
        (&hbuf[0][0][0])[i] = (_Float16)0.f;
        (&cbuf[0][0])[i]    = 0.f;
    }
    if (tid < Bc) Ls[tid] = seq_len[tid];
    __syncthreads();

    const int arow  = mtile * 16 + lcol;         // A-matrix row this lane feeds
    const int Lrow  = Ls[arow];
    const int kselA = (lane & 16) ? 16 : 0;      // A lane-half K offset
    const int kselB = (lane & 16) ? 16 : 0;      // B lane-half K offset

    _Float16 (*hcur)[HDc] = hbuf[0];
    _Float16 (*hnxt)[HDc] = hbuf[1];

    for (int step = 0; step < Tc; ++step) {
        int tx = step;
        if (dir) tx = (step < Lrow) ? (Lrow - 1 - step) : step;   // reversed-within-seq
        const _Float16* xrow = x + ((size_t)arow * Tc + tx) * din + kselA;

        #pragma unroll 1                     // serialize: one htile's state live at a time
        for (int ht = 0; ht < 4; ++ht) {
            const int htg = hgrp * 4 + ht;   // global h col-tile 0..15
            v8f a0 = {}, a1 = {}, a2 = {}, a3 = {};

            // per-gate weight row pointers (incremented, no per-load 64-bit math)
            const _Float16* wp0 = W + (size_t)(htg * 16 + lcol) * Kd + kselB;
            const _Float16* wp1 = wp0 + (size_t)256 * Kd;
            const _Float16* wp2 = wp0 + (size_t)512 * Kd;
            const _Float16* wp3 = wp0 + (size_t)768 * Kd;

            // ---- K part 1: input features from GLOBAL ----
            const _Float16* xp = xrow;
            for (int kc = 0; kc < din; kc += 32) {
                v8h lo = *(const v8h*)(xp);
                v8h hi = *(const v8h*)(xp + 8);
                xp += 32;
                v16h af = __builtin_shufflevector(lo, hi,
                    0,1,2,3,4,5,6,7,8,9,10,11,12,13,14,15);
                __builtin_prefetch((const void*)(wp0 + 32), 0, 1);
                v16h bf0 = *(const v16h*)wp0;  wp0 += 32;
                v16h bf1 = *(const v16h*)wp1;  wp1 += 32;
                a0 = __builtin_amdgcn_wmma_f32_16x16x32_f16(false, af, false, bf0, (short)0, a0, false, false);
                bf0 = *(const v16h*)wp2;       wp2 += 32;
                a1 = __builtin_amdgcn_wmma_f32_16x16x32_f16(false, af, false, bf1, (short)0, a1, false, false);
                bf1 = *(const v16h*)wp3;       wp3 += 32;
                a2 = __builtin_amdgcn_wmma_f32_16x16x32_f16(false, af, false, bf0, (short)0, a2, false, false);
                a3 = __builtin_amdgcn_wmma_f32_16x16x32_f16(false, af, false, bf1, (short)0, a3, false, false);
            }
            // ---- K part 2: recurrent features from LDS ----
            const _Float16* hp = &hcur[arow][kselA];
            for (int kc = 0; kc < HDc; kc += 32) {
                v8h lo = *(const v8h*)(hp);
                v8h hi = *(const v8h*)(hp + 8);
                hp += 32;
                v16h af = __builtin_shufflevector(lo, hi,
                    0,1,2,3,4,5,6,7,8,9,10,11,12,13,14,15);
                v16h bf0 = *(const v16h*)wp0;  wp0 += 32;
                v16h bf1 = *(const v16h*)wp1;  wp1 += 32;
                a0 = __builtin_amdgcn_wmma_f32_16x16x32_f16(false, af, false, bf0, (short)0, a0, false, false);
                bf0 = *(const v16h*)wp2;       wp2 += 32;
                a1 = __builtin_amdgcn_wmma_f32_16x16x32_f16(false, af, false, bf1, (short)0, a1, false, false);
                bf1 = *(const v16h*)wp3;       wp3 += 32;
                a2 = __builtin_amdgcn_wmma_f32_16x16x32_f16(false, af, false, bf0, (short)0, a2, false, false);
                a3 = __builtin_amdgcn_wmma_f32_16x16x32_f16(false, af, false, bf1, (short)0, a3, false, false);
            }

            // ---- elementwise LSTM cell for this htile ----
            const int hc = htg * 16 + lcol;
            const float bi = bs[0 * HDc + hc];
            const float bf = bs[1 * HDc + hc];
            const float bg = bs[2 * HDc + hc];
            const float bo = bs[3 * HDc + hc];
            #pragma unroll
            for (int r = 0; r < 8; ++r) {
                const int b  = mtile * 16 + lrow + r;
                const int Lb = Ls[b];
                const float i_ = sigf(a0[r] + bi);
                const float f_ = sigf(a1[r] + bf);
                const float g_ = tanh_fast(a2[r] + bg);
                const float o_ = sigf(a3[r] + bo);
                const float cold = cbuf[b][hc];      // lane-exclusive cell state
                const float c2 = f_ * cold + i_ * g_;
                const float h2 = o_ * tanh_fast(c2);
                const bool  m  = step < Lb;
                cbuf[b][hc] = m ? c2 : cold;
                const float hprev = (float)hcur[b][hc];
                hnxt[b][hc] = (_Float16)(m ? h2 : hprev);   // write NEXT buffer
                int tout = step;
                if (dir) tout = m ? (Lb - 1 - step) : step;
                out[((size_t)b * Tc + tout) * Hc + dir * HDc + hc] =
                    (_Float16)(m ? h2 : 0.f);
            }
        }
        __syncthreads();          // one barrier per step: hnxt/cbuf fully written
        _Float16 (*tmp)[HDc] = hcur; hcur = hnxt; hnxt = tmp;
    }
}

// ---------------- 4) LayerNorm + output projection (one wave per row) ----------------
__global__ void __launch_bounds__(256)
k_head(const _Float16* __restrict__ h, const float* __restrict__ gamma,
       const float* __restrict__ beta, const float* __restrict__ wo,
       const float* __restrict__ bo, float* __restrict__ feats)
{
    const int    lane = threadIdx.x & 31;
    const size_t row  = (size_t)blockIdx.x * 8 + (threadIdx.x >> 5);
    const _Float16* hr = h + row * Hc;

    float v[16];
    float s = 0.f;
    #pragma unroll
    for (int i = 0; i < 16; ++i) { v[i] = (float)hr[lane * 16 + i]; s += v[i]; }
    #pragma unroll
    for (int o = 16; o >= 1; o >>= 1) s += __shfl_xor(s, o, 32);
    const float mu = s * (1.f / 512.f);
    float q = 0.f;
    #pragma unroll
    for (int i = 0; i < 16; ++i) { const float d = v[i] - mu; q += d * d; }
    #pragma unroll
    for (int o = 16; o >= 1; o >>= 1) q += __shfl_xor(q, o, 32);
    const float rstd = rsqrtf(q * (1.f / 512.f) + 1e-5f);
    #pragma unroll
    for (int i = 0; i < 16; ++i)
        v[i] = (v[i] - mu) * rstd * gamma[lane * 16 + i] + beta[lane * 16 + i];

    for (int k = 0; k < Kc; ++k) {
        float p = 0.f;
        #pragma unroll
        for (int i = 0; i < 16; ++i) p += v[i] * wo[k * Hc + lane * 16 + i];
        #pragma unroll
        for (int o = 16; o >= 1; o >>= 1) p += __shfl_xor(p, o, 32);
        if (lane == 0) feats[row * Kc + k] = p + bo[k];
    }
}

// ---------------- 5) CRF forward + gold score + mean ----------------
__global__ void __launch_bounds__(128)
k_crf(const float* __restrict__ feats, const int* __restrict__ seq_len,
      const int* __restrict__ tags, const float* __restrict__ trans,
      float* __restrict__ outv)
{
    __shared__ float Tr[Kc * Kc];
    __shared__ float red[Bc];
    const int b = threadIdx.x;
    if (b < Kc * Kc) Tr[b] = trans[b];
    __syncthreads();

    float alpha[Kc];
    #pragma unroll
    for (int k = 0; k < Kc; ++k) alpha[k] = (k == BEGINt) ? 0.f : -10000.f;
    const int L = seq_len[b];
    float gold = 0.f;
    int   prev = BEGINt;

    for (int t = 0; t < Tc; ++t) {
        if (t >= L) break;                        // masked tail: alpha/gold frozen
        const float* em = feats + ((size_t)b * Tc + t) * Kc;
        float e[Kc];
        #pragma unroll
        for (int k = 0; k < Kc; ++k) e[k] = em[k];
        float na[Kc];
        #pragma unroll
        for (int j = 0; j < Kc; ++j) {
            float mx = -3.4e38f;
            #pragma unroll
            for (int k = 0; k < Kc; ++k) mx = fmaxf(mx, alpha[k] + Tr[j * Kc + k]);
            float ss = 0.f;
            #pragma unroll
            for (int k = 0; k < Kc; ++k) ss += __expf(alpha[k] + Tr[j * Kc + k] - mx);
            na[j] = e[j] + mx + __logf(ss);
        }
        #pragma unroll
        for (int j = 0; j < Kc; ++j) alpha[j] = na[j];
        const int tg = tags[b * Tc + t];
        gold += Tr[tg * Kc + prev] + e[tg];
        prev = tg;
    }
    gold += Tr[ENDt * Kc + prev];

    float mx = -3.4e38f;
    #pragma unroll
    for (int k = 0; k < Kc; ++k) mx = fmaxf(mx, alpha[k] + Tr[ENDt * Kc + k]);
    float ss = 0.f;
    #pragma unroll
    for (int k = 0; k < Kc; ++k) ss += __expf(alpha[k] + Tr[ENDt * Kc + k] - mx);
    const float fwd = mx + __logf(ss);

    red[b] = fwd - gold;
    __syncthreads();
    for (int s = 64; s >= 1; s >>= 1) {
        if (b < s) red[b] += red[b + s];
        __syncthreads();
    }
    if (b == 0) outv[0] = red[0] * (1.f / (float)Bc);
}

// ---------------- host launcher ----------------
extern "C" void kernel_launch(void* const* d_in, const int* in_sizes, int n_in,
                              void* d_out, int out_size, void* d_ws, size_t ws_size,
                              hipStream_t stream)
{
    (void)in_sizes; (void)n_in; (void)out_size; (void)ws_size;

    const int*   sentence = (const int*)  d_in[0];
    const int*   seq_len  = (const int*)  d_in[1];
    const int*   tags     = (const int*)  d_in[2];
    const float* embed    = (const float*)d_in[3];
    // lstm_params: f0, b0, f1, b1  each {w_ih, w_hh, b_ih, b_hh}
    const float* f0_wih = (const float*)d_in[4];
    const float* f0_whh = (const float*)d_in[5];
    const float* f0_bih = (const float*)d_in[6];
    const float* f0_bhh = (const float*)d_in[7];
    const float* b0_wih = (const float*)d_in[8];
    const float* b0_whh = (const float*)d_in[9];
    const float* b0_bih = (const float*)d_in[10];
    const float* b0_bhh = (const float*)d_in[11];
    const float* f1_wih = (const float*)d_in[12];
    const float* f1_whh = (const float*)d_in[13];
    const float* f1_bih = (const float*)d_in[14];
    const float* f1_bhh = (const float*)d_in[15];
    const float* b1_wih = (const float*)d_in[16];
    const float* b1_whh = (const float*)d_in[17];
    const float* b1_bih = (const float*)d_in[18];
    const float* b1_bhh = (const float*)d_in[19];
    const float* ln_g   = (const float*)d_in[20];
    const float* ln_b   = (const float*)d_in[21];
    const float* w_out  = (const float*)d_in[22];
    const float* b_out  = (const float*)d_in[23];
    const float* trans  = (const float*)d_in[24];

    char*  ws  = (char*)d_ws;
    size_t off = 0;
    auto alloc = [&](size_t bytes) -> void* {
        void* p = ws + off;
        off = (off + bytes + 255) & ~(size_t)255;
        return p;
    };

    _Float16* x16   = (_Float16*)alloc((size_t)Bc * Tc * Ec * 2);        // 32 MB
    _Float16* h0    = (_Float16*)alloc((size_t)Bc * Tc * Hc * 2);        // 64 MB
    _Float16* h1    = (_Float16*)alloc((size_t)Bc * Tc * Hc * 2);        // 64 MB
    _Float16* Wc0   = (_Float16*)alloc((size_t)2 * 1024 * (Ec + HDc) * 2);
    _Float16* Wc1   = (_Float16*)alloc((size_t)2 * 1024 * (Hc + HDc) * 2);
    float*    bias0 = (float*)   alloc((size_t)2 * 1024 * 4);
    float*    bias1 = (float*)   alloc((size_t)2 * 1024 * 4);
    float*    feats = (float*)   alloc((size_t)Bc * Tc * Kc * 4);        // 3 MB

    // 1) embedding gather + cast
    k_embed<<<Bc * Tc, 256, 0, stream>>>(sentence, embed, x16);

    // 2) pack weights (layer 0: Kd=512 ; layer 1: Kd=768), dir 1 offset = 1024*Kd
    k_prepw<<<1024, 256, 0, stream>>>(f0_wih, f0_whh, f0_bih, f0_bhh, Ec,
                                      Wc0,                             bias0);
    k_prepw<<<1024, 256, 0, stream>>>(b0_wih, b0_whh, b0_bih, b0_bhh, Ec,
                                      Wc0 + (size_t)1024 * (Ec + HDc), bias0 + 1024);
    k_prepw<<<1024, 256, 0, stream>>>(f1_wih, f1_whh, f1_bih, f1_bhh, Hc,
                                      Wc1,                             bias1);
    k_prepw<<<1024, 256, 0, stream>>>(b1_wih, b1_whh, b1_bih, b1_bhh, Hc,
                                      Wc1 + (size_t)1024 * (Hc + HDc), bias1 + 1024);

    // 3) recurrences: layer 0 then layer 1 (both directions concurrent, grid=2)
    k_lstm<<<2, 1024, 0, stream>>>(x16, Ec, Wc0, bias0, seq_len, h0);
    k_lstm<<<2, 1024, 0, stream>>>(h0,  Hc, Wc1, bias1, seq_len, h1);

    // 4) layernorm + projection to 12 tag scores
    k_head<<<(Bc * Tc) / 8, 256, 0, stream>>>(h1, ln_g, ln_b, w_out, b_out, feats);

    // 5) CRF forward/gold + mean -> scalar
    k_crf<<<1, Bc, 0, stream>>>(feats, seq_len, tags, trans, (float*)d_out);
}
